// Network_81003083203391
// MI455X (gfx1250) — matrix operand order repaired and checked by reading, not compiled, not run on hardware
//
#include <hip/hip_runtime.h>
#include <hip/hip_bf16.h>

// Problem constants
#define NB   65536   // batch
#define NG   6       // games
#define NIN  512
#define NF   256
#define NO   18

#define MT   32      // rows per block (two 16-row WMMA M-tiles)

typedef __attribute__((ext_vector_type(16))) __bf16 v16bf;
typedef __attribute__((ext_vector_type(8)))  __bf16 v8bf;
typedef __attribute__((ext_vector_type(8)))  float  v8f;

static __device__ __forceinline__ __bf16 f2bf(float f) {
    union { float f; unsigned u; } v; v.f = f;
    unsigned r = v.u + 0x7fffu + ((v.u >> 16) & 1u);   // round-to-nearest-even
    unsigned short h = (unsigned short)(r >> 16);
    return __builtin_bit_cast(__bf16, h);
}

// ---------------- counting sort (stable argsort of idx, G=6) ----------------

__global__ __launch_bounds__(256) void hist_kernel(const int* __restrict__ idx,
                                                   int* __restrict__ counts) {
    __shared__ int c[NG];
    if (threadIdx.x < NG) c[threadIdx.x] = 0;
    __syncthreads();
    int b = blockIdx.x * 256 + threadIdx.x;
    atomicAdd(&c[idx[b]], 1);
    __syncthreads();
    if (threadIdx.x < NG) counts[blockIdx.x * NG + threadIdx.x] = c[threadIdx.x];
}

__global__ void scan_kernel(const int* __restrict__ counts,
                            int* __restrict__ offsets) {
    if (threadIdx.x == 0 && blockIdx.x == 0) {
        int run = 0;
        for (int g = 0; g < NG; ++g)
            for (int blk = 0; blk < 256; ++blk) {
                offsets[blk * NG + g] = run;
                run += counts[blk * NG + g];
            }
    }
}

__global__ __launch_bounds__(256) void scatter_kernel(const int* __restrict__ idx,
                                                      const int* __restrict__ offsets,
                                                      int* __restrict__ perm) {
    __shared__ int sg[256];
    __shared__ int base[NG];
    int t = threadIdx.x;
    int b = blockIdx.x * 256 + t;
    int g = idx[b];
    sg[t] = g;
    if (t < NG) base[t] = offsets[blockIdx.x * NG + t];
    __syncthreads();
    int rank = 0;
    for (int j = 0; j < t; ++j) rank += (sg[j] == g);
    perm[base[g] + rank] = b;   // perm[sortedPos] = original row
}

// ---------------- weight pre-pack into WMMA B-fragment layout ----------------
// B fragment (32x16, bf16): lane l (0..15): N = l, K = 0..7,16..23 ; lanes 16..31: N = l-16, K = 8..15,24..31.
// Stored contiguously: frag f, lane l -> 16 bf16 at ((f*32 + l)*16).

__global__ __launch_bounds__(256) void pack_w1(const float* __restrict__ W1,
                                               __bf16* __restrict__ W1p) {
    int t = blockIdx.x * 256 + threadIdx.x;       // 6*256*32 = 49152 threads
    if (t >= NG * 256 * 32) return;
    int l  = t & 31;
    int f  = (t >> 5) & 255;                      // frag within game: f = ntile*16 + kk
    int g  = t >> 13;
    int kk = f & 15, ntile = f >> 4;
    int n  = ntile * 16 + (l & 15);
    int kb = kk * 32 + ((l < 16) ? 0 : 8);
    __bf16* dst = W1p + (size_t)t * 16;
    const float* src = W1 + (size_t)g * NIN * NF + n;
    #pragma unroll
    for (int j = 0; j < 16; ++j) dst[j] = f2bf(src[(size_t)(kb + j + ((j >= 8) ? 8 : 0)) * NF]);
}

__global__ __launch_bounds__(256) void pack_w2(const float* __restrict__ W2,
                                               __bf16* __restrict__ W2p) {
    int t = blockIdx.x * 256 + threadIdx.x;       // 128*32 = 4096 threads
    if (t >= 128 * 32) return;
    int l  = t & 31;
    int f  = t >> 5;                              // f = ntile*8 + kk
    int kk = f & 7, ntile = f >> 3;
    int n  = ntile * 16 + (l & 15);
    int kb = kk * 32 + ((l < 16) ? 0 : 8);
    __bf16* dst = W2p + (size_t)t * 16;
    const float* src = W2 + n;
    #pragma unroll
    for (int j = 0; j < 16; ++j) dst[j] = f2bf(src[(size_t)(kb + j + ((j >= 8) ? 8 : 0)) * NF]);
}

// ---------------- fused MoE MLP + head ----------------
// Block = 256 threads (8 waves), owns 32 consecutive sorted rows.
// Wave w: M-subtile mt = w>>2 (rows mt*16..mt*16+15), N-group wg = w&3 (cols wg*64..wg*64+63).

__global__ __launch_bounds__(256) void fused_kernel(
    const float* __restrict__ state, const int* __restrict__ action,
    const int* __restrict__ idx,
    const float* __restrict__ b1, const float* __restrict__ b2,
    const float* __restrict__ Wq, const float* __restrict__ bq,
    const int* __restrict__ perm,
    const __bf16* __restrict__ W1p, const __bf16* __restrict__ W2p,
    float* __restrict__ out)
{
    // Phase-aliased LDS: sA (32x512 bf16, 32KB) lives in phase 1; sHf (32x256 f32,
    // 32KB) reuses the same bytes in phase 3 (a __syncthreads separates them).
    __shared__ __align__(16) char smem[MT * NIN * 2 + MT * NF * 2];  // 48 KB
    __bf16* sA  = (__bf16*)smem;                 // [MT][NIN]
    float*  sHf = (float*)smem;                  // [MT][NF]
    __bf16* sH1 = (__bf16*)(smem + MT * NIN * 2); // [MT][NF]
    __shared__ int sPerm[MT];
    __shared__ int sG[MT];

    const int tid  = threadIdx.x;
    const int wave = tid >> 5;
    const int lane = tid & 31;
    const int b0   = blockIdx.x * MT;

    const int mt    = wave >> 2;        // 0/1: which 16-row M-subtile
    const int wg    = wave & 3;         // which 64-col N-group
    const int rbase = mt * 16;

    if (tid < MT) {
        int s = perm[b0 + tid];
        sPerm[tid] = s;
        sG[tid]    = idx[s];            // expert of this sorted row
    }
    __syncthreads();

    // Stage state tile -> LDS (f32 -> bf16), vectorized float4 reads.
    for (int e4 = tid; e4 < MT * (NIN / 4); e4 += 256) {
        int r  = e4 >> 7;
        int k4 = (e4 & 127) * 4;
        const float4 v = *(const float4*)(state + (size_t)sPerm[r] * NIN + k4);
        __bf16* d = &sA[r * NIN + k4];
        d[0] = f2bf(v.x); d[1] = f2bf(v.y); d[2] = f2bf(v.z); d[3] = f2bf(v.w);
    }
    __syncthreads();

    const int mrow = (lane < 16) ? 0 : 8;    // D-matrix row offset for this half-wave
    const int ncol = lane & 15;
    const int m    = lane & 15;              // A-matrix row for this lane

    float hsel[4][8];
    #pragma unroll
    for (int nt = 0; nt < 4; ++nt)
        #pragma unroll
        for (int v = 0; v < 8; ++v) hsel[nt][v] = 0.0f;

    // ---------- layer 1: [32x512] x [512x256], per-expert with row select ----------
    const int g_lo = sG[0], g_hi = sG[MT - 1];   // sorted => contiguous expert range
    for (int g = g_lo; g <= g_hi; ++g) {
        // presence within this wave's 16 rows (uniform within wave)
        bool present = false;
        #pragma unroll
        for (int r = 0; r < 16; ++r) present |= (sG[rbase + r] == g);
        if (!present) continue;

        v8f acc[4] = {v8f{}, v8f{}, v8f{}, v8f{}};
        for (int kk = 0; kk < 16; ++kk) {
            const int ks = kk * 32 + ((lane < 16) ? 0 : 8);
            const v8bf a0 = *(const v8bf*)&sA[(rbase + m) * NIN + ks];
            const v8bf a1 = *(const v8bf*)&sA[(rbase + m) * NIN + ks + 16];
            const v16bf a = __builtin_shufflevector(a0, a1,
                0,1,2,3,4,5,6,7,8,9,10,11,12,13,14,15);
            #pragma unroll
            for (int nt = 0; nt < 4; ++nt) {
                const int ntile = wg * 4 + nt;
                const v16bf bfrag = *(const v16bf*)(
                    W1p + ((size_t)(g * 256 + ntile * 16 + kk) * 32 + lane) * 16);
                acc[nt] = __builtin_amdgcn_wmma_f32_16x16x32_bf16(
                    false, a, false, bfrag, (short)0, acc[nt], false, false);
            }
        }
        #pragma unroll
        for (int nt = 0; nt < 4; ++nt) {
            const int n = (wg * 4 + nt) * 16 + ncol;
            const float bias = b1[g * NF + n];
            #pragma unroll
            for (int v = 0; v < 8; ++v) {
                float x = acc[nt][v] + bias;
                float s = 1.0f / (1.0f + __expf(-x));            // sigmoid
                hsel[nt][v] = (sG[rbase + v + mrow] == g) ? s : hsel[nt][v]; // branchless
            }
        }
    }

    // write h1 (bf16) to LDS
    #pragma unroll
    for (int nt = 0; nt < 4; ++nt) {
        const int n = (wg * 4 + nt) * 16 + ncol;
        #pragma unroll
        for (int v = 0; v < 8; ++v)
            sH1[(rbase + v + mrow) * NF + n] = f2bf(hsel[nt][v]);
    }
    __syncthreads();   // also separates last sA read from sHf writes (aliased)

    // ---------- layer 2: [32x256] x [256x256] ----------
    v8f acc2[4] = {v8f{}, v8f{}, v8f{}, v8f{}};
    for (int kk = 0; kk < 8; ++kk) {
        const int ks = kk * 32 + ((lane < 16) ? 0 : 8);
        const v8bf a0 = *(const v8bf*)&sH1[(rbase + m) * NF + ks];
        const v8bf a1 = *(const v8bf*)&sH1[(rbase + m) * NF + ks + 16];
        const v16bf a = __builtin_shufflevector(a0, a1,
            0,1,2,3,4,5,6,7,8,9,10,11,12,13,14,15);
        #pragma unroll
        for (int nt = 0; nt < 4; ++nt) {
            const int ntile = wg * 4 + nt;
            const v16bf bfrag = *(const v16bf*)(
                W2p + ((size_t)(ntile * 8 + kk) * 32 + lane) * 16);
            acc2[nt] = __builtin_amdgcn_wmma_f32_16x16x32_bf16(
                false, a, false, bfrag, (short)0, acc2[nt], false, false);
        }
    }
    __syncthreads();   // everyone past layer-2 reads of sH1 before sHf overwrite (aliased region is sA, but keep order strict)
    #pragma unroll
    for (int nt = 0; nt < 4; ++nt) {
        const int n = (wg * 4 + nt) * 16 + ncol;
        const float bias = b2[n];
        #pragma unroll
        for (int v = 0; v < 8; ++v) {
            float x = acc2[nt][v] + bias;
            sHf[(rbase + v + mrow) * NF + n] = fmaxf(x, 0.0f);   // relu
        }
    }
    __syncthreads();

    // ---------- fused head: q_out[b] = tanh( hf_row . Wq[idx[b]][:,action[b]] + bq ) ----------
    {
        const int r = tid >> 3;            // row 0..31 (output index b0+r)
        const int j = tid & 7;             // 8 lanes per row
        const int b = b0 + r;
        const int g2 = idx[b];             // ORIGINAL-order idx (reference semantics)
        const int a  = action[b];          // ORIGINAL-order action
        const float* wq = Wq + (size_t)g2 * NF * NO + a;
        float sum = 0.0f;
        #pragma unroll 4
        for (int k = j; k < NF; k += 8)
            sum += sHf[r * NF + k] * wq[(size_t)k * NO];
        sum += __shfl_down(sum, 4, 8);
        sum += __shfl_down(sum, 2, 8);
        sum += __shfl_down(sum, 1, 8);
        if (j == 0)
            out[b] = tanhf(sum + bq[g2 * NO + a]);
    }
}

// ---------------- host launcher ----------------

extern "C" void kernel_launch(void* const* d_in, const int* in_sizes, int n_in,
                              void* d_out, int out_size, void* d_ws, size_t ws_size,
                              hipStream_t stream) {
    const float* state  = (const float*)d_in[0];
    const int*   action = (const int*)d_in[1];
    const int*   idx    = (const int*)d_in[2];
    const float* W1     = (const float*)d_in[3];
    const float* b1     = (const float*)d_in[4];
    const float* W2     = (const float*)d_in[5];
    const float* b2     = (const float*)d_in[6];
    const float* Wq     = (const float*)d_in[7];
    const float* bq     = (const float*)d_in[8];
    float* out = (float*)d_out;

    char* ws = (char*)d_ws;
    int*    perm    = (int*)(ws + 0);                 // 65536*4      = 262144
    int*    counts  = (int*)(ws + 262144);            // 256*6*4      = 6144
    int*    offsets = (int*)(ws + 268288);            // 256*6*4      = 6144
    __bf16* W1p     = (__bf16*)(ws + 274432);         // 6*256*32*16*2 = 1572864
    __bf16* W2p     = (__bf16*)(ws + 1847296);        // 128*32*16*2   = 131072
    (void)in_sizes; (void)n_in; (void)out_size; (void)ws_size;

    hist_kernel   <<<256, 256, 0, stream>>>(idx, counts);
    scan_kernel   <<<1,   64,  0, stream>>>(counts, offsets);
    scatter_kernel<<<256, 256, 0, stream>>>(idx, offsets, perm);
    pack_w1       <<<192, 256, 0, stream>>>(W1, W1p);
    pack_w2       <<<16,  256, 0, stream>>>(W2, W2p);

    fused_kernel<<<NB / MT, 256, 0, stream>>>(
        state, action, idx, b1, b2, Wq, bq, perm, W1p, W2p, out);
}
